// GNN_34746285424883
// MI455X (gfx1250) — matrix operand order repaired and checked by reading, not compiled, not run on hardware
//
#include <hip/hip_runtime.h>
#include <limits>

// ---------------------------------------------------------------------------
// 2-layer GAT for MI455X / gfx1250.
//  - GEMMs: bf16 WMMA (fp32 accum), B-tile staged in LDS via the Tensor Data
//    Mover (TDM) when available, ds_load reads in the inner loop.
//  - Edge phase: CSR-by-destination built once (count/scan/scatter), then one
//    wave per (node, head) does segment-softmax + aggregation with zero
//    atomics in the hot loop; lanes = channels, shfl_xor wave reductions.
// ---------------------------------------------------------------------------

typedef __attribute__((ext_vector_type(16))) __bf16       v16bf;
typedef __attribute__((ext_vector_type(8)))  float        v8f;
typedef __attribute__((ext_vector_type(4)))  unsigned int v4u;
typedef __attribute__((ext_vector_type(8)))  int          v8i;
typedef __attribute__((ext_vector_type(4)))  int          v4i;

#define NEG_SLOPE 0.2f

// ---------------------------------------------------------------------------
// GEMM: C[n x NCOLS] = A[n x K] * B[K x NCOLS]   (fp32 in, bf16 WMMA, fp32 acc)
// Block = 256 threads = 8 waves; all waves share one 16-column tile of B
// (tileN = blockIdx.x % NTN), each wave owns one 16-row tile of A/C.
// B tile (K x 16) is staged into LDS (TDM if available), reads via ds_load.
// WMMA VGPR layouts per CDNA5 ISA 7.12.2.
// ---------------------------------------------------------------------------
template <int K, int NCOLS, int NTN>
__global__ void gat_gemm_wmma(const float* __restrict__ A,
                              const float* __restrict__ B,
                              float* __restrict__ C, int n) {
  __shared__ float Bs[K * 16];

  const int tileN = blockIdx.x % NTN;
  const int tileM = (blockIdx.x / NTN) * 8 + (threadIdx.x >> 5);
  const int lane  = threadIdx.x & 31;
  const int g     = lane >> 4;   // half-wave group
  const int ml    = lane & 15;

#if __has_builtin(__builtin_amdgcn_tensor_load_to_lds)
  // --- Stage B tile with the Tensor Data Mover (wave 0 issues; TDM ignores
  // EXEC and signals TENSORcnt once when the whole tile is in LDS). ---
  if (threadIdx.x < 32) {
    const unsigned long long gaddr =
        (unsigned long long)(size_t)B + (unsigned long long)tileN * 16ull * 4ull;
    const unsigned int ldsa = (unsigned int)(size_t)&Bs[0];  // low 32 bits = LDS offset
    const int td0 = NCOLS - tileN * 16;  // valid cols from tile start (OOB reads -> 0)
    // D# group 0: count=1 | lds_addr | global_addr[56:0] | type=2
    v4u g0 = {1u, ldsa, (unsigned int)gaddr,
              (unsigned int)(((gaddr >> 32) & 0x1FFFFFFull) | (2u << 30))};
    // D# group 1: data_size=4B; tensor_dim0=td0, tensor_dim1=K;
    //             tile_dim0=16, tile_dim1=K, tile_dim2=0; dim0_stride=NCOLS.
    v8i g1 = {(int)(2u << 16),
              (int)((unsigned)(td0 & 0xFFFF) << 16),
              (int)(((td0 >> 16) & 0xFFFF) | ((K & 0xFFFF) << 16)),
              (int)(((K >> 16) & 0xFFFF) | (16 << 16)),
              (int)(K & 0xFFFF),
              (int)NCOLS,
              0, 0};
    v4i z4 = {0, 0, 0, 0};
    v8i z8 = {0, 0, 0, 0, 0, 0, 0, 0};
    __builtin_amdgcn_tensor_load_to_lds(g0, g1, z4, z4, z8, 0);
    __builtin_amdgcn_s_wait_tensorcnt(0);
  }
#else
  for (int idx = threadIdx.x; idx < K * 16; idx += blockDim.x) {
    const int kk = idx >> 4, j = idx & 15;
    const int col = tileN * 16 + j;
    Bs[idx] = (col < NCOLS) ? B[(size_t)kk * NCOLS + col] : 0.f;
  }
#endif
  __syncthreads();

  // Clamp instead of predicating A loads: duplicated tail rows compute garbage
  // that is masked at the store, no exec-mask branches in the inner loop.
  const int rowA = min(tileM * 16 + ml, n - 1);
  const int colB = tileN * 16 + ml;

  v8f acc = {};
#pragma unroll
  for (int k0 = 0; k0 < K; k0 += 32) {
    v16bf a, b;
#pragma unroll
    for (int v = 0; v < 8; ++v) {
      const int kk = k0 + ((v < 4) ? (g * 8 + 2 * v) : (16 + g * 8 + 2 * (v - 4)));
      float2 f = *(const float2*)(A + (size_t)rowA * K + kk);
      a[2 * v]     = (__bf16)f.x;
      a[2 * v + 1] = (__bf16)f.y;
    }
#pragma unroll
    for (int i = 0; i < 16; ++i) {
      b[i] = (__bf16)Bs[(k0 + ((i < 8) ? (g * 8 + i) : (16 + g * 8 + (i - 8)))) * 16 + ml];  // ds_load
    }
    acc = __builtin_amdgcn_wmma_f32_16x16x32_bf16(
        /*neg_a=*/false, a, /*neg_b=*/false, b,
        /*c_mod=*/(short)0, acc, /*reuse_a=*/false, /*reuse_b=*/false);
  }
#pragma unroll
  for (int r = 0; r < 8; ++r) {
    const int row = tileM * 16 + r + 8 * g;
    if (row < n && colB < NCOLS) C[(size_t)row * NCOLS + colB] = acc[r];
  }
}

// ---------------------------------------------------------------------------
// Per-(node,head) attention coefficients.
// ---------------------------------------------------------------------------
__global__ void alpha_k(const float* __restrict__ h, const float* __restrict__ a_s,
                        const float* __restrict__ a_d, float* __restrict__ as,
                        float* __restrict__ ad, int n, int H, int C) {
  int i = blockIdx.x * blockDim.x + threadIdx.x;
  if (i >= n * H) return;
  int node = i / H, hh = i % H;
  const float* hp = h + (size_t)node * H * C + (size_t)hh * C;
  float s1 = 0.f, s2 = 0.f;
  for (int c = 0; c < C; ++c) {
    float v = hp[c];
    s1 += v * a_s[hh * C + c];
    s2 += v * a_d[hh * C + c];
  }
  as[i] = s1;
  ad[i] = s2;
}

// ---------------------------------------------------------------------------
// CSR-by-destination build. Edges [0,E) from edge_index, [E,E+n) = self loops.
// ---------------------------------------------------------------------------
__global__ void fill_i32_k(int* __restrict__ p, int v, int n) {
  int i = blockIdx.x * blockDim.x + threadIdx.x;
  if (i < n) p[i] = v;
}

__global__ void copy_i32_k(const int* __restrict__ s, int* __restrict__ d, int n) {
  int i = blockIdx.x * blockDim.x + threadIdx.x;
  if (i < n) d[i] = s[i];
}

__global__ void count_k(const long long* __restrict__ dstI, int* __restrict__ cnt,
                        int E, int n) {
  int i = blockIdx.x * blockDim.x + threadIdx.x;
  if (i >= E + n) return;
  int d = (i < E) ? (int)dstI[i] : (i - E);
  atomicAdd(&cnt[d], 1);
}

// Single-block exclusive scan: each of 1024 threads owns a contiguous chunk,
// Hillis-Steele scan of the 1024 chunk sums, then local sweep. ~20 barriers.
__global__ void scan_k(const int* __restrict__ cnt, int* __restrict__ off, int n) {
  __shared__ int ssum[1024];
  const int tid = threadIdx.x;
  const int chunk = (n + 1023) / 1024;
  const int lo = tid * chunk;
  const int hi = min(lo + chunk, n);
  int s = 0;
  for (int i = lo; i < hi; ++i) s += cnt[i];
  ssum[tid] = s;
  __syncthreads();
  for (int d = 1; d < 1024; d <<= 1) {
    int t = (tid >= d) ? ssum[tid - d] : 0;
    __syncthreads();
    ssum[tid] += t;
    __syncthreads();
  }
  int run = ssum[tid] - s;  // exclusive prefix of this chunk
  for (int i = lo; i < hi; ++i) {
    off[i] = run;
    run += cnt[i];
  }
  if (tid == 1023) off[n] = ssum[1023];
}

__global__ void scatter_k(const long long* __restrict__ srcI,
                          const long long* __restrict__ dstI,
                          int* __restrict__ cursor, int* __restrict__ elist,
                          int E, int n) {
  int i = blockIdx.x * blockDim.x + threadIdx.x;
  if (i >= E + n) return;
  int s, d;
  if (i < E) { s = (int)srcI[i]; d = (int)dstI[i]; } else { s = d = i - E; }
  int pos = atomicAdd(&cursor[d], 1);
  elist[pos] = s;
}

// ---------------------------------------------------------------------------
// One wave per (node, head): segment-softmax + weighted aggregation, fused
// bias (+ optional ELU). Lanes = channels; no atomics.
// ---------------------------------------------------------------------------
template <int H, int C, bool ELU>
__global__ void gat_aggregate_k(const int* __restrict__ off, const int* __restrict__ elist,
                                const float* __restrict__ as, const float* __restrict__ ad,
                                const float* __restrict__ h, const float* __restrict__ bias,
                                float* __restrict__ out, int n) {
  const int gtid = blockIdx.x * blockDim.x + threadIdx.x;
  const int seg = gtid >> 5;
  const int lane = threadIdx.x & 31;
  if (seg >= n * H) return;  // wave-uniform
  const int node = seg / H;
  const int hh = seg % H;
  const int beg = off[node];
  const int end = off[node + 1];  // >= beg+1 (self loop always present)
  const float adv = ad[node * H + hh];

  // Pass 1: segment max (edge-parallel across lanes).
  float mx = -3.4e38f;
  for (int j = beg + lane; j < end; j += 32) {
    float v = as[elist[j] * H + hh] + adv;
    v = v > 0.f ? v : NEG_SLOPE * v;
    mx = fmaxf(mx, v);
  }
#pragma unroll
  for (int d = 16; d > 0; d >>= 1) mx = fmaxf(mx, __shfl_xor(mx, d, 32));

  // Pass 2: partition sum.
  float sum = 0.f;
  for (int j = beg + lane; j < end; j += 32) {
    float v = as[elist[j] * H + hh] + adv;
    v = v > 0.f ? v : NEG_SLOPE * v;
    sum += __expf(v - mx);
  }
#pragma unroll
  for (int d = 16; d > 0; d >>= 1) sum += __shfl_xor(sum, d, 32);

  // Pass 3: weighted feature aggregation, lanes own channels.
  constexpr int RC = (C + 31) / 32;
  float acc[RC];
#pragma unroll
  for (int r = 0; r < RC; ++r) acc[r] = 0.f;

  for (int j = beg; j < end; ++j) {
    const int s = elist[j];  // uniform per iteration
    float v = as[s * H + hh] + adv;
    v = v > 0.f ? v : NEG_SLOPE * v;
    const float w = __expf(v - mx);
    const float* hp = h + (size_t)s * H * C + (size_t)hh * C;
#pragma unroll
    for (int r = 0; r < RC; ++r) {
      const int c = lane + 32 * r;
      if (c < C) acc[r] += w * hp[c];
    }
  }

  const float inv = 1.f / (sum + 1e-16f);
  float* op = out + (size_t)node * H * C + (size_t)hh * C;
#pragma unroll
  for (int r = 0; r < RC; ++r) {
    const int c = lane + 32 * r;
    if (c < C) {
      float v = acc[r] * inv + bias[hh * C + c];
      if (ELU) v = v > 0.f ? v : (__expf(v) - 1.f);
      op[c] = v;
    }
  }
}

__global__ void logsoftmax_k(const float* __restrict__ in, float* __restrict__ out,
                             int n, int C) {
  int node = blockIdx.x * blockDim.x + threadIdx.x;
  if (node >= n) return;
  const float* p = in + (size_t)node * C;
  float mx = p[0];
  for (int c = 1; c < C; ++c) mx = fmaxf(mx, p[c]);
  float sum = 0.f;
  for (int c = 0; c < C; ++c) sum += __expf(p[c] - mx);
  float lse = mx + __logf(sum);
  float* o = out + (size_t)node * C;
  for (int c = 0; c < C; ++c) o[c] = p[c] - lse;
}

// ---------------------------------------------------------------------------
extern "C" void kernel_launch(void* const* d_in, const int* in_sizes, int n_in,
                              void* d_out, int out_size, void* d_ws, size_t ws_size,
                              hipStream_t stream) {
  (void)n_in; (void)out_size; (void)ws_size;
  const float*     x      = (const float*)d_in[0];
  const long long* ei     = (const long long*)d_in[1];  // int64 [2, E]
  const float*     W1     = (const float*)d_in[2];
  const float*     a_src1 = (const float*)d_in[3];
  const float*     a_dst1 = (const float*)d_in[4];
  const float*     b1     = (const float*)d_in[5];
  const float*     W2     = (const float*)d_in[6];
  const float*     a_src2 = (const float*)d_in[7];
  const float*     a_dst2 = (const float*)d_in[8];
  const float*     b2     = (const float*)d_in[9];

  const int F = 128, H1 = 8, C1 = 32, HC1 = 256, C2 = 40;
  const int N = in_sizes[0] / F;
  const int E = in_sizes[1] / 2;
  const long long* srcI = ei;
  const long long* dstI = ei + E;

  // Workspace layout.
  char* wsb = (char*)d_ws;
  float* h1  = (float*)wsb; wsb += (size_t)N * HC1 * 4;  // layer-1 WMMA output
  float* h1e = (float*)wsb; wsb += (size_t)N * HC1 * 4;  // layer-1 GAT output (post-ELU)
  float* as1 = (float*)wsb; wsb += (size_t)N * H1 * 4;
  float* ad1 = (float*)wsb; wsb += (size_t)N * H1 * 4;
  float* h2  = (float*)wsb; wsb += (size_t)N * C2 * 4;
  float* lg2 = (float*)wsb; wsb += (size_t)N * C2 * 4;   // layer-2 logits
  float* as2 = (float*)wsb; wsb += (size_t)N * 4;
  float* ad2 = (float*)wsb; wsb += (size_t)N * 4;
  int* cnt   = (int*)wsb;   wsb += (size_t)N * 4;
  int* off   = (int*)wsb;   wsb += (size_t)(N + 1) * 4;
  int* cur   = (int*)wsb;   wsb += (size_t)N * 4;
  int* elist = (int*)wsb;   wsb += (size_t)(E + N) * 4;

  const int BT = 256;  // 8 waves per block (wave32)
  auto nb = [&](long long threads) { return (int)((threads + BT - 1) / BT); };

  // ---- CSR by destination (shared by both layers) ----
  fill_i32_k<<<nb(N), BT, 0, stream>>>(cnt, 0, N);
  count_k<<<nb((long long)E + N), BT, 0, stream>>>(dstI, cnt, E, N);
  scan_k<<<1, 1024, 0, stream>>>(cnt, off, N);
  copy_i32_k<<<nb(N), BT, 0, stream>>>(off, cur, N);
  scatter_k<<<nb((long long)E + N), BT, 0, stream>>>(srcI, dstI, cur, elist, E, N);

  // ---- Layer 1 ----
  {
    const int tilesM = (N + 15) / 16;
    const int blocks = ((tilesM + 7) / 8) * 16;  // NTN = 16 column tiles
    gat_gemm_wmma<128, 256, 16><<<blocks, BT, 0, stream>>>(x, W1, h1, N);
  }
  alpha_k<<<nb((long long)N * H1), BT, 0, stream>>>(h1, a_src1, a_dst1, as1, ad1, N, H1, C1);
  gat_aggregate_k<8, 32, true>
      <<<nb((long long)N * H1 * 32), BT, 0, stream>>>(off, elist, as1, ad1, h1, b1, h1e, N);

  // ---- Layer 2 ----
  {
    const int tilesM = (N + 15) / 16;
    const int blocks = ((tilesM + 7) / 8) * 3;   // 40 cols -> 3 tiles of 16
    gat_gemm_wmma<256, 40, 3><<<blocks, BT, 0, stream>>>(h1e, W2, h2, N);
  }
  alpha_k<<<nb((long long)N), BT, 0, stream>>>(h2, a_src2, a_dst2, as2, ad2, N, 1, C2);
  gat_aggregate_k<1, 40, false>
      <<<nb((long long)N * 32), BT, 0, stream>>>(off, elist, as2, ad2, h2, b2, lg2, N);

  logsoftmax_k<<<nb(N), BT, 0, stream>>>(lg2, (float*)d_out, N, C2);
}
